// AutoQVLALinearW8A16_56581899157846
// MI455X (gfx1250) — matrix operand order repaired and checked by reading, not compiled, not run on hardware
//
#include <hip/hip_runtime.h>
#include <cstdint>

// W8A16 linear: y[m,n] = scale[n] * sum_k x[m,k]*q[n,k] + bias[n]
// M=8, K=8192, N=28672. HBM-bound on the 235MB int8 weight stream (~10us @ 23.3TB/s).

typedef _Float16 h16 __attribute__((ext_vector_type(16)));
typedef _Float16 h8  __attribute__((ext_vector_type(8)));
typedef _Float16 h2  __attribute__((ext_vector_type(2)));
typedef float    f8  __attribute__((ext_vector_type(8)));
typedef uint32_t u4  __attribute__((ext_vector_type(4)));

#define K_DIM 8192
#define N_DIM 28672
#define M_DIM 8
#define WAVES 8
#define KSEG  (K_DIM / WAVES)   // 1024 K per wave -> 32 WMMA steps per slab

union AVec { h16 v; h8 half8[2]; };
union BVec { h16 v; uint32_t w[8]; };

// 16 int8 weights -> 16 f16, order-preserving, 20 VALU:
//   xw  = raw ^ 0x80808080                      (bias bytes)
//   v_perm vs 0x64646464 builds halves 0x6400|(b^0x80) == 1152 + b
//   v_pk_add_f16 -1152
__device__ __forceinline__ void cvt16(const u4 raw, BVec& b, uint32_t c64) {
    h2 mb; mb[0] = mb[1] = (_Float16)(-1152.0f);
    #pragma unroll
    for (int j = 0; j < 4; ++j) {
        uint32_t xw = raw[j] ^ 0x80808080u;
        uint32_t lo = __builtin_amdgcn_perm(c64, xw, 0x04010400u); // {0x64,b1',0x64,b0'}
        uint32_t hi = __builtin_amdgcn_perm(c64, xw, 0x04030402u); // {0x64,b3',0x64,b2'}
        h2 tl = __builtin_bit_cast(h2, lo) + mb;                   // v_pk_add_f16
        h2 th = __builtin_bit_cast(h2, hi) + mb;
        b.w[2 * j]     = __builtin_bit_cast(uint32_t, tl);
        b.w[2 * j + 1] = __builtin_bit_cast(uint32_t, th);
    }
}

__global__ __launch_bounds__(256) void w8a16_wmma_kernel(
    const _Float16* __restrict__ x,       // [M, K] fp16
    const int8_t*   __restrict__ qw,      // [N, K] int8
    const _Float16* __restrict__ scales,  // [N, 1] fp16
    const _Float16* __restrict__ bias,    // [N]    fp16
    float* __restrict__ out)              // [M, N] fp32
{
    __shared__ float red[WAVES][2][32][8];

    const int tid   = threadIdx.x;
    const int wave  = tid >> 5;
    const int lane  = tid & 31;
    const int h     = lane >> 4;          // lane half: K sub-range select
    const int m     = lane & 15;          // A row / B column-within-slab
    const int nbase = blockIdx.x * 32;    // two 16-wide N slabs per block-wave
    const int kbase = wave * KSEG;
    const uint32_t c64 = 0x64646464u;

    f8 acc0 = {};
    f8 acc1 = {};

    // B: lane streams 16 contiguous int8 along K of its weight row per slab
    const int8_t* wrow0 = qw + (size_t)(nbase + m)      * K_DIM + kbase + h * 16;
    const int8_t* wrow1 = qw + (size_t)(nbase + 16 + m) * K_DIM + kbase + h * 16;
    // A: branchless — lanes with m>=8 duplicate row m&7; D rows 8..15 are
    // duplicates of rows 0..7 and are simply not stored.
    const _Float16* arow = x + (size_t)(m & 7) * K_DIM + kbase + h * 16;

    for (int kb = 0; kb < KSEG; kb += 128) {
        // one cacheline of per-lane weight stream per outer iter, ~8 iters ahead
        __builtin_prefetch(wrow0 + kb + 1024, 0, 3);
        __builtin_prefetch(wrow1 + kb + 1024, 0, 3);
        #pragma unroll
        for (int j = 0; j < 4; ++j) {
            const int k0 = kb + 32 * j;

            AVec a;                                       // 16x32 f16 A tile
            a.half8[0] = *(const h8*)(arow + k0);         // K = k0 +     h*16 ..+7
            a.half8[1] = *(const h8*)(arow + k0 + 8);     // K = k0 + 8 + h*16 ..+7

            u4 r0 = *(const u4*)(wrow0 + k0);             // K = k0 + h*16 ..+15
            u4 r1 = *(const u4*)(wrow1 + k0);
            BVec b0, b1;
            cvt16(r0, b0, c64);
            cvt16(r1, b1, c64);

            acc0 = __builtin_amdgcn_wmma_f32_16x16x32_f16(
                false, a.v, false, b0.v, (short)0, acc0, false, false);
            acc1 = __builtin_amdgcn_wmma_f32_16x16x32_f16(
                false, a.v, false, b1.v, (short)0, acc1, false, false);
        }
    }

    // ---- cross-wave K reduction through LDS ----
    #pragma unroll
    for (int s = 0; s < 8; ++s) {
        red[wave][0][lane][s] = acc0[s];
        red[wave][1][lane][s] = acc1[s];
    }
    __syncthreads();

    const int slot = tid >> 5;            // C/D VGPR index 0..7
    const int l2   = tid & 31;
    const int mrow = slot + 8 * (l2 >> 4);  // rows 8..15 are duplicates -> skipped
    if (mrow < M_DIM) {
        const int ncol0 = nbase + (l2 & 15);
        const int ncol1 = ncol0 + 16;
        float s0 = 0.0f, s1 = 0.0f;
        #pragma unroll
        for (int w = 0; w < WAVES; ++w) {
            s0 += red[w][0][l2][slot];
            s1 += red[w][1][l2][slot];
        }
        out[(size_t)mrow * N_DIM + ncol0] = s0 * (float)scales[ncol0] + (float)bias[ncol0];
        out[(size_t)mrow * N_DIM + ncol1] = s1 * (float)scales[ncol1] + (float)bias[ncol1];
    }
}

extern "C" void kernel_launch(void* const* d_in, const int* in_sizes, int n_in,
                              void* d_out, int out_size, void* d_ws, size_t ws_size,
                              hipStream_t stream) {
    (void)in_sizes; (void)n_in; (void)out_size; (void)d_ws; (void)ws_size;
    const _Float16* x   = (const _Float16*)d_in[0];
    const int8_t*   qw  = (const int8_t*)d_in[1];
    const _Float16* sc  = (const _Float16*)d_in[2];
    const _Float16* bs  = (const _Float16*)d_in[3];
    float*          out = (float*)d_out;

    dim3 grid(N_DIM / 32);   // 896 blocks, each covering 32 output columns
    dim3 block(256);         // 8 waves; each wave owns K/8 for both slabs
    hipLaunchKernelGGL(w8a16_wmma_kernel, grid, block, 0, stream,
                       x, qw, sc, bs, out);
}
